// GAT_5471788335739
// MI455X (gfx1250) — matrix operand order repaired and checked by reading, not compiled, not run on hardware
//
#include <hip/hip_runtime.h>
#include <math.h>

// ---------------------------------------------------------------------------
// GATv2 2-layer GNN for MI455X (gfx1250, wave32, WMMA).
// Dense node transforms use v_wmma_f32_16x16x32_f16 (f16 A/B, f32 accum).
// LDS tiles are staged in *fragment-ready* layout so each lane loads its
// v16h WMMA operand with 2x ds_load_b128 (no per-half gathers/packs).
// Edge phase keeps xl/xr fp32 (102.4MB total -> resident in 192MB L2).
// ---------------------------------------------------------------------------

typedef __attribute__((ext_vector_type(16))) _Float16 v16h;
typedef __attribute__((ext_vector_type(8)))  float    v8f;

#define HC_DIM 128
#define ED_DIM 5
#define NEG_SLOPE 0.2f

__device__ __forceinline__ void atomicMaxF(float* addr, float v) {
  // int/uint reinterpret trick -> single hardware atomic, no CAS loop
  if (v >= 0.0f) atomicMax((int*)addr, __float_as_int(v));
  else           atomicMin((unsigned int*)addr, __float_as_uint(v));
}

// ---------------------------------------------------------------------------
// out[Nrows, M] = act(A[Nrows, K] @ W[K, M] + bias)
// KP = K padded to multiple of 32, MP = M padded to multiple of 16 (<=128).
// One block = 16 rows; 8 waves, wave w -> output columns [16w, 16w+16).
//
// Fragment-ready LDS layouts (derived from CDNA5 ISA 7.12.2):
//   A frag slot: As[(chunk*32 + lane)*16 + i]            (32B / lane / chunk)
//     holds A[row0 + lane%16][chunk*32 + (i/8)*16 + (lane/16)*8 + i%8]
//   B frag slot: Ws[((chunk*NTILE + tile)*32 + lane)*16 + i]
//     holds W[chunk*32 + (i/8)*16 + (lane/16)*8 + i%8][tile*16 + lane%16]
// ---------------------------------------------------------------------------
template<int KP, int MP, int TANH>
__global__ void gemm16_wmma(const float* __restrict__ A,
                            const float* __restrict__ W,
                            const float* __restrict__ bias,
                            float* __restrict__ out,
                            int Nrows, int K, int M) {
  constexpr int NCHUNK = KP / 32;
  constexpr int NTILE  = MP / 16;
  __shared__ __align__(16) _Float16 Ws[KP * MP];
  __shared__ __align__(16) _Float16 As[16 * KP];
  const int t = threadIdx.x;
  const int row0 = blockIdx.x * 16;

  // ---- stage W into fragment-ready layout, packed pair stores ----
  unsigned int* WsU = (unsigned int*)Ws;
  for (int wi = t; wi < (KP * MP) / 2; wi += 256) {
    int fi   = wi << 1;
    int i0   = fi & 15;                 // even
    int lane = (fi >> 4) & 31;
    int tc   = fi >> 9;                 // / (16*32)
    int tile = tc % NTILE;
    int chnk = tc / NTILE;
    int k0 = chnk * 32 + ((i0 >> 3) << 4) + ((lane >> 4) << 3) + (i0 & 7);
    int m  = (tile << 4) + (lane & 15);
    float f0 = (k0     < K && m < M) ? W[k0 * M + m]       : 0.0f;
    float f1 = (k0 + 1 < K && m < M) ? W[(k0 + 1) * M + m] : 0.0f;
    union { _Float16 h[2]; unsigned int u; } p;
    p.h[0] = (_Float16)f0; p.h[1] = (_Float16)f1;
    WsU[wi] = p.u;
  }
  // ---- stage A tile into fragment-ready layout ----
  unsigned int* AsU = (unsigned int*)As;
  for (int wi = t; wi < (16 * KP) / 2; wi += 256) {
    int fi   = wi << 1;
    int i0   = fi & 15;                 // even
    int lane = (fi >> 4) & 31;
    int chnk = fi >> 9;
    int half = lane >> 4;
    int r    = lane & 15;
    int k0 = chnk * 32 + ((i0 >> 3) << 4) + (half << 3) + (i0 & 7);
    int row = row0 + r;
    float f0 = (row < Nrows && k0     < K) ? A[(size_t)row * K + k0]     : 0.0f;
    float f1 = (row < Nrows && k0 + 1 < K) ? A[(size_t)row * K + k0 + 1] : 0.0f;
    union { _Float16 h[2]; unsigned int u; } p;
    p.h[0] = (_Float16)f0; p.h[1] = (_Float16)f1;
    AsU[wi] = p.u;
  }
  __syncthreads();

  const int wave = t >> 5;     // 8 waves
  const int lane = t & 31;
  const int half = lane >> 4;
  const int l16  = lane & 15;
  if (wave * 16 >= MP) return;

  v8f acc = {};
#pragma unroll
  for (int chnk = 0; chnk < NCHUNK; ++chnk) {
    v16h a = *(const v16h*)(As + ((chnk * 32 + lane) << 4));
    v16h b = *(const v16h*)(Ws + (((chnk * NTILE + wave) * 32 + lane) << 4));
    acc = __builtin_amdgcn_wmma_f32_16x16x32_f16(false, a, false, b,
                                                 (short)0, acc, false, false);
  }

  const int n = wave * 16 + l16;
  if (n < M) {
    float bv = bias[n];
#pragma unroll
    for (int j = 0; j < 8; ++j) {
      int row = row0 + j + (half << 3);
      if (row < Nrows) {
        float v = acc[j] + bv;
        if (TANH) v = tanhf(v);
        out[(size_t)row * M + n] = v;
      }
    }
  }
}

// ---------------------------------------------------------------------------
// Per-conv init: accum[N*128]=0 ; lmax[N*2]=-inf ; den[N*2]=0
// ---------------------------------------------------------------------------
__global__ void conv_init(float* __restrict__ lmax, float* __restrict__ den,
                          float* __restrict__ accum, int N) {
  int i = blockIdx.x * blockDim.x + threadIdx.x;
  if (i < N * HC_DIM) accum[i] = 0.0f;
  if (i < N * 2) { lmax[i] = __int_as_float(0xff800000); den[i] = 0.0f; }
}

// ---------------------------------------------------------------------------
// Edge pass 1: logits[e,h] = sum_c leaky(xl[s]+xr[d]+ea@We)[h,c] * att[h,c]
// e_emb (E x 128 = 410MB if materialized) is recomputed on the fly from the
// 5-wide edge_attr — trades ~1 GF of VALU for ~800MB of HBM traffic.
// ---------------------------------------------------------------------------
__global__ void edge_logits(const int* __restrict__ src, const int* __restrict__ dst,
                            const float* __restrict__ ea,
                            const float* __restrict__ We,   // [5,128]
                            const float* __restrict__ att,  // [2,64] flat
                            const float* __restrict__ xl,
                            const float* __restrict__ xr,
                            float* __restrict__ logits,     // [E,2]
                            float* __restrict__ lmax,       // [N,2]
                            int E) {
  int e = blockIdx.x * blockDim.x + threadIdx.x;
  if (e >= E) return;
  int s = src[e], d = dst[e];
  float a0 = ea[(size_t)e*ED_DIM+0], a1 = ea[(size_t)e*ED_DIM+1],
        a2 = ea[(size_t)e*ED_DIM+2], a3 = ea[(size_t)e*ED_DIM+3],
        a4 = ea[(size_t)e*ED_DIM+4];
  const float4* pl = (const float4*)(xl + (size_t)s * HC_DIM);
  const float4* pr = (const float4*)(xr + (size_t)d * HC_DIM);
  const float4* W4 = (const float4*)We;   // row stride 32 float4
  const float4* A4 = (const float4*)att;  // 32 float4
  float lg0 = 0.0f, lg1 = 0.0f;
#pragma unroll 4
  for (int q = 0; q < 32; ++q) {
    float4 lv = pl[q], rv = pr[q];
    float4 w0 = W4[q], w1 = W4[32+q], w2 = W4[64+q], w3 = W4[96+q], w4 = W4[128+q];
    float4 aw = A4[q];
    float acc = 0.0f;
#define GAT_COMP(mem)                                                        \
    { float mm = lv.mem + rv.mem + a0*w0.mem + a1*w1.mem + a2*w2.mem         \
                 + a3*w3.mem + a4*w4.mem;                                    \
      mm = (mm > 0.0f) ? mm : NEG_SLOPE * mm;                                \
      acc += mm * aw.mem; }
    GAT_COMP(x) GAT_COMP(y) GAT_COMP(z) GAT_COMP(w)
#undef GAT_COMP
    if (q < 16) lg0 += acc; else lg1 += acc;
  }
  logits[(size_t)e*2 + 0] = lg0;
  logits[(size_t)e*2 + 1] = lg1;
  atomicMaxF(&lmax[(size_t)d*2 + 0], lg0);
  atomicMaxF(&lmax[(size_t)d*2 + 1], lg1);
}

// ---------------------------------------------------------------------------
// Edge pass 2: ex = exp(logit - lmax[dst]); den[dst] += ex (in-place)
// ---------------------------------------------------------------------------
__global__ void edge_exp(const int* __restrict__ dst,
                         float* __restrict__ logits,
                         const float* __restrict__ lmax,
                         float* __restrict__ den, int E) {
  int e = blockIdx.x * blockDim.x + threadIdx.x;
  if (e >= E) return;
  int d = dst[e];
#pragma unroll
  for (int h = 0; h < 2; ++h) {
    float ex = __expf(logits[(size_t)e*2 + h] - lmax[(size_t)d*2 + h]);
    logits[(size_t)e*2 + h] = ex;
    atomicAdd(&den[(size_t)d*2 + h], ex);
  }
}

// ---------------------------------------------------------------------------
// Edge pass 3: accum[dst] += (ex/den[dst]) * xl[src]  (L2-resident atomics)
// ---------------------------------------------------------------------------
__global__ void edge_aggr(const int* __restrict__ src, const int* __restrict__ dst,
                          const float* __restrict__ exs,
                          const float* __restrict__ den,
                          const float* __restrict__ xl,
                          float* __restrict__ accum, int E) {
  int e = blockIdx.x * blockDim.x + threadIdx.x;
  if (e >= E) return;
  int s = src[e], d = dst[e];
  float al0 = exs[(size_t)e*2 + 0] / (den[(size_t)d*2 + 0] + 1e-16f);
  float al1 = exs[(size_t)e*2 + 1] / (den[(size_t)d*2 + 1] + 1e-16f);
  const float* pl = xl + (size_t)s * HC_DIM;
  float* po = accum + (size_t)d * HC_DIM;
#pragma unroll 8
  for (int c = 0; c < 64; ++c)   atomicAdd(&po[c], al0 * pl[c]);
#pragma unroll 8
  for (int c = 64; c < 128; ++c) atomicAdd(&po[c], al1 * pl[c]);
}

// ---------------------------------------------------------------------------
// buf = (tanh?)(buf + bias[c])
// ---------------------------------------------------------------------------
template<int TANH>
__global__ void node_finalize(float* __restrict__ buf,
                              const float* __restrict__ bias, int N) {
  int i = blockIdx.x * blockDim.x + threadIdx.x;
  if (i >= N * HC_DIM) return;
  float v = buf[i] + bias[i % HC_DIM];
  if (TANH) v = tanhf(v);
  buf[i] = v;
}

// ---------------------------------------------------------------------------
extern "C" void kernel_launch(void* const* d_in, const int* in_sizes, int n_in,
                              void* d_out, int out_size, void* d_ws, size_t ws_size,
                              hipStream_t stream) {
  const float* x    = (const float*)d_in[0];
  const int*   ei   = (const int*)  d_in[1];
  const float* ea   = (const float*)d_in[2];
  const float* W0   = (const float*)d_in[3];
  const float* b0   = (const float*)d_in[4];
  const float* Wl1  = (const float*)d_in[5];
  const float* bl1  = (const float*)d_in[6];
  const float* Wr1  = (const float*)d_in[7];
  const float* br1  = (const float*)d_in[8];
  const float* We1  = (const float*)d_in[9];
  const float* att1 = (const float*)d_in[10];
  const float* bias1= (const float*)d_in[11];
  const float* Wl2  = (const float*)d_in[12];
  const float* bl2  = (const float*)d_in[13];
  const float* Wr2  = (const float*)d_in[14];
  const float* br2  = (const float*)d_in[15];
  const float* We2  = (const float*)d_in[16];
  const float* att2 = (const float*)d_in[17];
  const float* bias2= (const float*)d_in[18];
  (void)n_in; (void)out_size; (void)ws_size;

  const int N = in_sizes[0] / 128;   // 100000
  const int E = in_sizes[1] / 2;     // 800000
  const int* src = ei;               // edge_index[0,:]
  const int* dst = ei + E;           // edge_index[1,:]

  // ---- workspace carve-out (256B aligned) ----
  char* ws = (char*)d_ws;
  size_t off = 0;
  auto carve = [&](size_t bytes) {
    void* p = ws + off;
    off += (bytes + 255) & ~(size_t)255;
    return p;
  };
  float* h1   = (float*)carve((size_t)N * 8 * sizeof(float));        // lin1 out
  float* hbuf = (float*)carve((size_t)N * HC_DIM * sizeof(float));   // conv1 out
  float* xl   = (float*)carve((size_t)N * HC_DIM * sizeof(float));
  float* xr   = (float*)carve((size_t)N * HC_DIM * sizeof(float));
  float* elog = (float*)carve((size_t)E * 2 * sizeof(float));        // logits/ex
  float* lmax = (float*)carve((size_t)N * 2 * sizeof(float));
  float* den  = (float*)carve((size_t)N * 2 * sizeof(float));
  float* outp = (float*)d_out;

  dim3 blk(256);
  int gRow16 = (N + 15) / 16;
  int gE     = (E + 255) / 256;
  int gNHC   = (N * HC_DIM + 255) / 256;

  // ---- lin1: h1 = tanh(x @ W0 + b0)   [K=128, M=8] ----
  gemm16_wmma<128, 16, 1><<<gRow16, blk, 0, stream>>>(x, W0, b0, h1, N, 128, 8);

  // ---- conv1 (K=8 -> 128) ----
  gemm16_wmma<32, 128, 0><<<gRow16, blk, 0, stream>>>(h1, Wl1, bl1, xl, N, 8, HC_DIM);
  gemm16_wmma<32, 128, 0><<<gRow16, blk, 0, stream>>>(h1, Wr1, br1, xr, N, 8, HC_DIM);
  conv_init  <<<gNHC, blk, 0, stream>>>(lmax, den, hbuf, N);
  edge_logits<<<gE,   blk, 0, stream>>>(src, dst, ea, We1, att1, xl, xr, elog, lmax, E);
  edge_exp   <<<gE,   blk, 0, stream>>>(dst, elog, lmax, den, E);
  edge_aggr  <<<gE,   blk, 0, stream>>>(src, dst, elog, den, xl, hbuf, E);
  node_finalize<0><<<gNHC, blk, 0, stream>>>(hbuf, bias1, N);

  // ---- conv2 (K=128 -> 128) + final tanh ----
  gemm16_wmma<128, 128, 0><<<gRow16, blk, 0, stream>>>(hbuf, Wl2, bl2, xl, N, HC_DIM, HC_DIM);
  gemm16_wmma<128, 128, 0><<<gRow16, blk, 0, stream>>>(hbuf, Wr2, br2, xr, N, HC_DIM, HC_DIM);
  conv_init  <<<gNHC, blk, 0, stream>>>(lmax, den, outp, N);
  edge_logits<<<gE,   blk, 0, stream>>>(src, dst, ea, We2, att2, xl, xr, elog, lmax, E);
  edge_exp   <<<gE,   blk, 0, stream>>>(dst, elog, lmax, den, E);
  edge_aggr  <<<gE,   blk, 0, stream>>>(src, dst, elog, den, xl, outp, E);
  node_finalize<1><<<gNHC, blk, 0, stream>>>(outp, bias2, N);
}
